// WindowAttention_65678639891268
// MI455X (gfx1250) — matrix operand order repaired and checked by reading, not compiled, not run on hardware
//
#include <hip/hip_runtime.h>
#include <hip/hip_bf16.h>

// ---------------------------------------------------------------------------
// Types for CDNA5 WMMA
// ---------------------------------------------------------------------------
typedef __attribute__((ext_vector_type(16))) __bf16 v16bf;
typedef __attribute__((ext_vector_type(8)))  float  v8f;

union BfFrag {
    v16bf v;
    uint4 q[2];
    unsigned short s[16];
};

__device__ __forceinline__ unsigned short f32_to_bf16(float f) {
    union { float f; unsigned int u; } x;
    x.f = f;
    unsigned int u = x.u;
    u += 0x7fffu + ((u >> 16) & 1u);   // round-to-nearest-even
    return (unsigned short)(u >> 16);
}

// ---------------------------------------------------------------------------
// Problem constants
// ---------------------------------------------------------------------------
#define DIMC   512
#define HEADS  8
#define HDIM   64              // per-head dim
#define NTOK   64              // tokens per window
#define NWIN   2048            // windows (32 lat * 64 lon)
#define MROWS  (NWIN * NTOK)   // 131072
#define QKVN   (3 * DIMC)      // 1536
#define PER_LAT 225            // (2*8-1)^2 bias slab per latitude row

// ---------------------------------------------------------------------------
// fp32 -> bf16 conversion (vectorized, grid-stride)
// ---------------------------------------------------------------------------
__global__ void cvt_f32_bf16(const float* __restrict__ src,
                             unsigned short* __restrict__ dst,
                             long long n4) {
    long long i = (long long)blockIdx.x * blockDim.x + threadIdx.x;
    long long stride = (long long)gridDim.x * blockDim.x;
    for (; i < n4; i += stride) {
        float4 f = reinterpret_cast<const float4*>(src)[i];
        ushort4 o;
        o.x = f32_to_bf16(f.x);
        o.y = f32_to_bf16(f.y);
        o.z = f32_to_bf16(f.z);
        o.w = f32_to_bf16(f.w);
        reinterpret_cast<ushort4*>(dst)[i] = o;
    }
}

// ---------------------------------------------------------------------------
// bf16 GEMM: C[m,n] = sum_k A[m,k] * W[n,k] + bias[n]
//   A: (M x K) bf16 row-major, W: (N x K) bf16 row-major
//   Tile 128(M) x 128(N), 256 threads = 8 waves (4x2), K step 32.
//   Software pipelined: next tile's global loads overlap current WMMAs.
// ---------------------------------------------------------------------------
template<int OUT_BF16>
__global__ __launch_bounds__(256)
void gemm_bf16_wmma(const unsigned short* __restrict__ A,
                    const unsigned short* __restrict__ W,
                    const float* __restrict__ bias,
                    float* __restrict__ Cf,
                    unsigned short* __restrict__ Cb,
                    int M, int N, int K) {
    __shared__ __align__(16) unsigned short lds_a[128][40]; // pad 32->40 halves
    __shared__ __align__(16) unsigned short lds_b[128][40];

    const int tid  = threadIdx.x;
    const int lane = tid & 31;
    const int wv   = tid >> 5;        // 0..7
    const int wm   = wv >> 1;         // 0..3 : 32-row slab
    const int wn   = wv & 1;          // 0..1 : 64-col slab
    const int lr   = lane & 15;
    const int lh   = lane >> 4;

    const int mBase = blockIdx.y * 128;
    const int nBase = blockIdx.x * 128;

    v8f acc[2][4];
    #pragma unroll
    for (int a = 0; a < 2; ++a)
        #pragma unroll
        for (int b = 0; b < 4; ++b)
            acc[a][b] = (v8f){};

    const int lrow = tid >> 1;         // 0..127
    const int lcol = (tid & 1) * 16;   // 0 or 16 (halves)

    const unsigned short* aRow = &A[(size_t)(mBase + lrow) * K + lcol];
    const unsigned short* wRow = &W[(size_t)(nBase + lrow) * K + lcol];

    // prologue: load first K-tile
    uint4 ra0 = *reinterpret_cast<const uint4*>(aRow);
    uint4 ra1 = *reinterpret_cast<const uint4*>(aRow + 8);
    uint4 rb0 = *reinterpret_cast<const uint4*>(wRow);
    uint4 rb1 = *reinterpret_cast<const uint4*>(wRow + 8);

    for (int k0 = 0; k0 < K; k0 += 32) {
        __syncthreads();
        *reinterpret_cast<uint4*>(&lds_a[lrow][lcol])     = ra0;
        *reinterpret_cast<uint4*>(&lds_a[lrow][lcol + 8]) = ra1;
        *reinterpret_cast<uint4*>(&lds_b[lrow][lcol])     = rb0;
        *reinterpret_cast<uint4*>(&lds_b[lrow][lcol + 8]) = rb1;
        __syncthreads();

        // issue next tile's global loads; they overlap the WMMAs below
        if (k0 + 32 < K) {
            ra0 = *reinterpret_cast<const uint4*>(aRow + k0 + 32);
            ra1 = *reinterpret_cast<const uint4*>(aRow + k0 + 40);
            rb0 = *reinterpret_cast<const uint4*>(wRow + k0 + 32);
            rb1 = *reinterpret_cast<const uint4*>(wRow + k0 + 40);
        }
        if (k0 + 64 < K) {
            __builtin_prefetch(aRow + k0 + 64, 0, 1);  // global_prefetch_b8
            __builtin_prefetch(wRow + k0 + 64, 0, 1);
        }

        // A fragments (16x32 bf16): lane-half lh -> K {0..7,16..23} / {8..15,24..31}
        BfFrag afr[2];
        #pragma unroll
        for (int a = 0; a < 2; ++a) {
            const int row = wm * 32 + a * 16 + lr;
            afr[a].q[0] = *reinterpret_cast<const uint4*>(&lds_a[row][lh * 8]);
            afr[a].q[1] = *reinterpret_cast<const uint4*>(&lds_a[row][16 + lh * 8]);
        }
        // B fragments (32x16 bf16): lane-half lh -> K {0..7,16..23} / {8..15,24..31}
        BfFrag bfr[4];
        #pragma unroll
        for (int b = 0; b < 4; ++b) {
            const int col = wn * 64 + b * 16 + lr;
            bfr[b].q[0] = *reinterpret_cast<const uint4*>(&lds_b[col][lh * 8]);
            bfr[b].q[1] = *reinterpret_cast<const uint4*>(&lds_b[col][16 + lh * 8]);
        }
        #pragma unroll
        for (int a = 0; a < 2; ++a)
            #pragma unroll
            for (int b = 0; b < 4; ++b)
                acc[a][b] = __builtin_amdgcn_wmma_f32_16x16x32_bf16(
                    false, afr[a].v, false, bfr[b].v,
                    (short)0, acc[a][b], false, false);
    }

    // Epilogue: C/D layout -> VGPR j holds row (lh*8 + j), lane lo = col
    #pragma unroll
    for (int a = 0; a < 2; ++a) {
        #pragma unroll
        for (int b = 0; b < 4; ++b) {
            #pragma unroll
            for (int v = 0; v < 8; ++v) {
                const int row = mBase + wm * 32 + a * 16 + lh * 8 + v;
                const int col = nBase + wn * 64 + b * 16 + lr;
                const float r = acc[a][b][v] + bias[col];
                if (OUT_BF16)
                    Cb[(size_t)row * N + col] = f32_to_bf16(r);
                else
                    Cf[(size_t)row * N + col] = r;
            }
        }
    }
}

// ---------------------------------------------------------------------------
// Fused window attention: one workgroup per (window b, head h).
//   qkv: (B*N, 1536) bf16, layout [q(512) | k(512) | v(512)], col h*64+d.
//   S = (Q K^T)*scale + earth bias (index computed analytically, no rel_idx
//   gather), softmax fully in registers via 16-lane shfl_xor reductions,
//   P written bf16 to LDS, O = P V with V^T staged in LDS.
// 128 threads = 4 waves; wave w owns rows [16w, 16w+16).
// ---------------------------------------------------------------------------
__global__ __launch_bounds__(128)
void win_attn_wmma(const unsigned short* __restrict__ qkv,
                   const float* __restrict__ bias_table,
                   unsigned short* __restrict__ attn) {
    __shared__ __align__(16) unsigned short s_p[NTOK][72];   // probs bf16
    __shared__ __align__(16) unsigned short s_vt[HDIM][72];  // V^T bf16

    const int blk  = blockIdx.x;
    const int b    = blk >> 3;        // window
    const int h    = blk & 7;         // head
    const int tid  = threadIdx.x;
    const int lane = tid & 31;
    const int wv   = tid >> 5;        // 0..3
    const int lr   = lane & 15;
    const int lh   = lane >> 4;
    const int r0   = wv * 16;         // row slab for this wave

    const size_t rowBase = (size_t)b * NTOK;   // first token row of window
    const int    qOff = h * HDIM;
    const int    kOff = DIMC + h * HDIM;
    const int    vOff = 2 * DIMC + h * HDIM;

    // ---- Phase 0: stage V^T into LDS (b128 global reads, b16 transposed stores)
    #pragma unroll
    for (int g = 0; g < 4; ++g) {
        const int grp = tid + g * 128;    // 0..511 groups of 8 elements
        const int j   = grp >> 3;         // token 0..63
        const int d0  = (grp & 7) * 8;    // head-dim 0..56
        uint4 vv = *reinterpret_cast<const uint4*>(
            &qkv[(rowBase + j) * QKVN + vOff + d0]);
        const unsigned short* vs = reinterpret_cast<const unsigned short*>(&vv);
        #pragma unroll
        for (int e = 0; e < 8; ++e) s_vt[d0 + e][j] = vs[e];
    }

    // ---- Phase 1: S = Q K^T (WMMA, straight from global bf16) ----
    v8f acc[4];
    #pragma unroll
    for (int t = 0; t < 4; ++t) acc[t] = (v8f){};

    #pragma unroll
    for (int d0 = 0; d0 < HDIM; d0 += 32) {
        BfFrag afr;
        {
            const int row = r0 + lr;
            const unsigned short* qp = &qkv[(rowBase + row) * QKVN + qOff + d0];
            afr.q[0] = *reinterpret_cast<const uint4*>(qp + lh * 8);
            afr.q[1] = *reinterpret_cast<const uint4*>(qp + 16 + lh * 8);
        }
        #pragma unroll
        for (int t = 0; t < 4; ++t) {
            BfFrag bfr;
            const int col = t * 16 + lr;  // key token
            const unsigned short* kp = &qkv[(rowBase + col) * QKVN + kOff + d0];
            bfr.q[0] = *reinterpret_cast<const uint4*>(kp + lh * 8);
            bfr.q[1] = *reinterpret_cast<const uint4*>(kp + 16 + lh * 8);
            acc[t] = __builtin_amdgcn_wmma_f32_16x16x32_bf16(
                false, afr.v, false, bfr.v, (short)0, acc[t], false, false);
        }
    }

    // ---- Phase 2: scale + analytic earth bias + register softmax ----
    // Row i lives at fixed (lh, v); its 64 columns are spread over
    // lanes lr=0..15 x tiles t=0..3  ->  reduce with shfl_xor {1,2,4,8}.
    const float scale = 0.125f;  // 64^-0.5
    const int   lat   = b >> 6;  // MLON = 64
    #pragma unroll
    for (int v = 0; v < 8; ++v) {
        const int i  = r0 + lh * 8 + v;
        const int ih = i >> 3, iw = i & 7;
        float sv[4];
        #pragma unroll
        for (int t = 0; t < 4; ++t) {
            const int j  = t * 16 + lr;
            const int jh = j >> 3, jw = j & 7;
            const int idx = lat * PER_LAT + (ih - jh + 7) * 15 + (iw - jw + 7);
            sv[t] = acc[t][v] * scale + bias_table[(size_t)idx * HEADS + h];
        }
        float mx = fmaxf(fmaxf(sv[0], sv[1]), fmaxf(sv[2], sv[3]));
        #pragma unroll
        for (int d = 1; d < 16; d <<= 1) mx = fmaxf(mx, __shfl_xor(mx, d, 32));
        float sum = 0.f;
        #pragma unroll
        for (int t = 0; t < 4; ++t) { sv[t] = __expf(sv[t] - mx); sum += sv[t]; }
        #pragma unroll
        for (int d = 1; d < 16; d <<= 1) sum += __shfl_xor(sum, d, 32);
        const float inv = 1.0f / sum;
        #pragma unroll
        for (int t = 0; t < 4; ++t)
            s_p[i][t * 16 + lr] = f32_to_bf16(sv[t] * inv);
    }
    __syncthreads();   // publishes s_p and s_vt to all waves

    // ---- Phase 3: O = P V (WMMA from LDS) ----
    v8f out[4];
    #pragma unroll
    for (int t = 0; t < 4; ++t) out[t] = (v8f){};

    #pragma unroll
    for (int j0 = 0; j0 < NTOK; j0 += 32) {
        BfFrag afr;
        {
            const int row = r0 + lr;
            afr.q[0] = *reinterpret_cast<const uint4*>(&s_p[row][j0 + lh * 8]);
            afr.q[1] = *reinterpret_cast<const uint4*>(&s_p[row][j0 + 16 + lh * 8]);
        }
        #pragma unroll
        for (int t = 0; t < 4; ++t) {
            BfFrag bfr;
            const int d = t * 16 + lr;
            bfr.q[0] = *reinterpret_cast<const uint4*>(&s_vt[d][j0 + lh * 8]);
            bfr.q[1] = *reinterpret_cast<const uint4*>(&s_vt[d][j0 + 16 + lh * 8]);
            out[t] = __builtin_amdgcn_wmma_f32_16x16x32_bf16(
                false, afr.v, false, bfr.v, (short)0, out[t], false, false);
        }
    }

    // store O as bf16 in (B*N, 512) layout, col = h*64 + d
    #pragma unroll
    for (int t = 0; t < 4; ++t) {
        #pragma unroll
        for (int v = 0; v < 8; ++v) {
            const int i = r0 + lh * 8 + v;
            const int d = t * 16 + lr;
            attn[(rowBase + i) * DIMC + h * HDIM + d] = f32_to_bf16(out[t][v]);
        }
    }
}

// ---------------------------------------------------------------------------
// Launch
// ---------------------------------------------------------------------------
extern "C" void kernel_launch(void* const* d_in, const int* in_sizes, int n_in,
                              void* d_out, int out_size, void* d_ws, size_t ws_size,
                              hipStream_t stream) {
    const float* x          = (const float*)d_in[0];
    const float* w_qkv      = (const float*)d_in[1];
    const float* b_qkv      = (const float*)d_in[2];
    const float* w_proj     = (const float*)d_in[3];
    const float* b_proj     = (const float*)d_in[4];
    const float* bias_table = (const float*)d_in[5];
    (void)d_in[6];  // rel_idx reproduced analytically in-kernel
    float* out = (float*)d_out;

    // workspace layout (bf16 halves)
    unsigned short* x_bf     = (unsigned short*)d_ws;               // 131072*512
    unsigned short* wqkv_bf  = x_bf    + (size_t)MROWS * DIMC;      // 1536*512
    unsigned short* wproj_bf = wqkv_bf + (size_t)QKVN * DIMC;       // 512*512
    unsigned short* qkv_bf   = wproj_bf + (size_t)DIMC * DIMC;      // 131072*1536
    unsigned short* attn_bf  = qkv_bf  + (size_t)MROWS * QKVN;      // 131072*512

    // 1) fp32 -> bf16 conversions
    {
        long long n4 = (long long)MROWS * DIMC / 4;
        cvt_f32_bf16<<<(int)((n4 + 255) / 256), 256, 0, stream>>>(x, x_bf, n4);
    }
    {
        long long n4 = (long long)QKVN * DIMC / 4;
        cvt_f32_bf16<<<(int)((n4 + 255) / 256), 256, 0, stream>>>(w_qkv, wqkv_bf, n4);
    }
    {
        long long n4 = (long long)DIMC * DIMC / 4;
        cvt_f32_bf16<<<(int)((n4 + 255) / 256), 256, 0, stream>>>(w_proj, wproj_bf, n4);
    }

    // 2) QKV projection: (131072 x 512) @ (512 x 1536) + b_qkv -> bf16
    {
        dim3 grid(QKVN / 128, MROWS / 128);
        gemm_bf16_wmma<1><<<grid, 256, 0, stream>>>(
            x_bf, wqkv_bf, b_qkv, nullptr, qkv_bf, MROWS, QKVN, DIMC);
    }

    // 3) fused window attention: one block per (window, head)
    {
        win_attn_wmma<<<NWIN * HEADS, 128, 0, stream>>>(
            qkv_bf, bias_table, attn_bf);
    }

    // 4) output projection: (131072 x 512) @ (512 x 512) + b_proj -> fp32 out
    {
        dim3 grid(DIMC / 128, MROWS / 128);
        gemm_bf16_wmma<0><<<grid, 256, 0, stream>>>(
            attn_bf, wproj_bf, b_proj, out, nullptr, MROWS, DIMC, DIMC);
    }
}
